// SimpleSelfAtt_3135326126157
// MI455X (gfx1250) — matrix operand order repaired and checked by reading, not compiled, not run on hardware
//
#include <hip/hip_runtime.h>

typedef __attribute__((ext_vector_type(16))) _Float16 v16h;
typedef __attribute__((ext_vector_type(8)))  _Float16 v8h;
typedef __attribute__((ext_vector_type(4)))  _Float16 h4;
typedef __attribute__((ext_vector_type(8)))  float    v8f;

#define B_    32
#define H_    16
#define N_    1024
#define D_    128
#define NT    256     // threads per block = 8 wave32
#define TROWS 128     // rows of t staged in LDS per tile
#define ROWH  136     // padded row stride in halves (+16B) to break LDS bank conflicts

// ---------------- kernel 1: padd[b] = first zero index in padding_t[b,:] (else N)
__global__ void k_padd(const int* __restrict__ pad, int* __restrict__ padd) {
  __shared__ int m;
  const int b = blockIdx.x;
  if (threadIdx.x == 0) m = N_;
  __syncthreads();
  for (int n = threadIdx.x; n < N_; n += NT)
    if (pad[b * N_ + n] == 0) atomicMin(&m, n);
  __syncthreads();
  if (threadIdx.x == 0) padd[b] = m;
}

// ---------------- kernel 2: score1[bh] = sum_{n,d} t1[bh,n,d] * w1[d]
__global__ void k_score1(const float* __restrict__ t1, const float* __restrict__ cw,
                         float* __restrict__ score1) {
  const int bh  = blockIdx.x;
  const int tid = threadIdx.x;
  const float4* p  = (const float4*)(t1 + (size_t)bh * N_ * D_);
  // element (float4) index i = tid + k*NT; since NT*4 % D_ == 0, d-chunk is loop-invariant
  const float4  w1 = ((const float4*)cw)[tid & (D_ / 4 - 1)];
  float acc = 0.f;
  for (int i = tid; i < N_ * D_ / 4; i += NT) {
    float4 v = p[i];
    acc += v.x * w1.x + v.y * w1.y + v.z * w1.z + v.w * w1.w;
  }
  for (int o = 16; o > 0; o >>= 1) acc += __shfl_down(acc, o, 32);
  __shared__ float part[NT / 32];
  const int lane = tid & 31, wave = tid >> 5;
  if (lane == 0) part[wave] = acc;
  __syncthreads();
  if (tid == 0) {
    float v = 0.f;
    #pragma unroll
    for (int i = 0; i < NT / 32; ++i) v += part[i];
    score1[bh] = v;
  }
}

// ---------------- kernel 3: WMMA scores + masked softmax + out = t*(1+attn)
__global__ void k_attn(const float* __restrict__ t,  const float* __restrict__ cw,
                       const float* __restrict__ cb, const int* __restrict__ padd,
                       const float* __restrict__ score1, float* __restrict__ out) {
  __shared__ _Float16 tile[TROWS * ROWH];   // ~34 KB f16 staging of a 128x128 slab
  __shared__ _Float16 w2h[D_];
  __shared__ float    scores[N_];
  __shared__ float    red[NT / 32];

  const int bh   = blockIdx.x;
  const int b    = bh / H_;
  const int tid  = threadIdx.x;
  const int lane = tid & 31;
  const int wave = tid >> 5;

  const float* slab  = t   + (size_t)bh * N_ * D_;
  float*       oslab = out + (size_t)bh * N_ * D_;

  if (tid < D_) w2h[tid] = (_Float16)cw[D_ + tid];
  __syncthreads();

  // B fragments (32x16 f16, w2 chunk replicated across all 16 columns).
  // Lane l holds column l%16, rows K = 16*(l/16)+0..15 of the 32-row chunk.
  v16h bfrag[4];
  #pragma unroll
  for (int kc = 0; kc < 4; ++kc) {
    const int off = kc * 32 + (lane >> 4) * 16;
    #pragma unroll
    for (int i = 0; i < 16; ++i) bfrag[kc][i] = w2h[off + i];
  }

  const int m     = lane & 15;          // A-matrix row within 16-row tile
  const int kbase = (lane >> 4) * 16;   // A-matrix K sub-offset (contiguous k = 16h + s, same order as B)

  for (int tile0 = 0; tile0 < N_; tile0 += TROWS) {
    // ---- stage 128x128 f32 -> f16 LDS, fully coalesced float4 reads
    const float4* src = (const float4*)(slab + (size_t)tile0 * D_);
    for (int i = tid; i < TROWS * D_ / 4; i += NT) {
      float4 v = src[i];
      const int r = i >> 5, c4 = i & 31;
      h4 hv = {(_Float16)v.x, (_Float16)v.y, (_Float16)v.z, (_Float16)v.w};
      *(h4*)&tile[r * ROWH + c4 * 4] = hv;
    }
    __syncthreads();

    // ---- each wave: 16 rows, 4 WMMA k-steps cover D=128
    const int rowbase = wave * 16;
    const _Float16* rowp = &tile[(rowbase + m) * ROWH];
    v8f c = {};
    #pragma unroll
    for (int kc = 0; kc < 4; ++kc) {
      const _Float16* pa = rowp + kc * 32 + kbase;
      v8h lo = *(const v8h*)(pa);
      v8h hi = *(const v8h*)(pa + 8);
      v16h a;
      #pragma unroll
      for (int i = 0; i < 8; ++i) { a[i] = lo[i]; a[8 + i] = hi[i]; }
      c = __builtin_amdgcn_wmma_f32_16x16x32_f16(false, a, false, bfrag[kc],
                                                 (short)0, c, false, false);
    }
    // every D-column equals the score; column 0 lanes (l%16==0) extract 8 rows each
    if ((lane & 15) == 0) {
      const int mb = (lane >> 4) * 8;
      #pragma unroll
      for (int r = 0; r < 8; ++r) scores[tile0 + rowbase + mb + r] = c[r];
    }
    __syncthreads();
  }

  // ---- masked softmax over n (in LDS)
  const float s1  = score1[bh] + cb[0];
  const int   pdd = padd[b];

  float lmax = -INFINITY;
  for (int n = tid; n < N_; n += NT) {
    const float sc = (n >= pdd) ? -INFINITY : (scores[n] + s1);
    scores[n] = sc;
    lmax = fmaxf(lmax, sc);
  }
  for (int o = 16; o > 0; o >>= 1) lmax = fmaxf(lmax, __shfl_down(lmax, o, 32));
  if (lane == 0) red[wave] = lmax;
  __syncthreads();
  if (tid == 0) {
    float v = red[0];
    #pragma unroll
    for (int i = 1; i < NT / 32; ++i) v = fmaxf(v, red[i]);
    red[0] = v;
  }
  __syncthreads();
  const float gmax = red[0];
  const bool  any  = (gmax > -INFINITY);
  __syncthreads();

  float lsum = 0.f;
  for (int n = tid; n < N_; n += NT) {
    const float e = (any && n < pdd) ? __expf(scores[n] - gmax) : 0.f;
    scores[n] = e;
    lsum += e;
  }
  for (int o = 16; o > 0; o >>= 1) lsum += __shfl_down(lsum, o, 32);
  __syncthreads();
  if (lane == 0) red[wave] = lsum;
  __syncthreads();
  if (tid == 0) {
    float v = 0.f;
    #pragma unroll
    for (int i = 0; i < NT / 32; ++i) v += red[i];
    red[0] = (v > 0.f) ? 1.f / v : 0.f;
  }
  __syncthreads();
  const float inv = red[0];

  // ---- out = t * (1 + attn); re-read of slab should hit L2 (512 KB working set)
  for (int i = tid; i < N_ * D_ / 4; i += NT) {
    const int n = i >> 5;
    const float s = 1.f + scores[n] * inv;   // scores[n]==0 for masked n
    float4 v = ((const float4*)slab)[i];
    float4 o = {v.x * s, v.y * s, v.z * s, v.w * s};
    ((float4*)oslab)[i] = o;
  }
}

extern "C" void kernel_launch(void* const* d_in, const int* in_sizes, int n_in,
                              void* d_out, int out_size, void* d_ws, size_t ws_size,
                              hipStream_t stream) {
  const float* t1  = (const float*)d_in[0];   // transformer_t_1 (B,H,N,D) f32
  const float* t   = (const float*)d_in[1];   // transformer_t   (B,H,N,D) f32
  const int*   pad = (const int*)  d_in[2];   // padding_t (B,N) i32
  const float* cw  = (const float*)d_in[4];   // concat_w (1,2D) f32
  const float* cb  = (const float*)d_in[5];   // concat_b (1,) f32
  float* out = (float*)d_out;

  int*   padd   = (int*)d_ws;
  float* score1 = (float*)((char*)d_ws + 256);  // B*H floats

  k_padd  <<<B_,       NT, 0, stream>>>(pad, padd);
  k_score1<<<B_ * H_,  NT, 0, stream>>>(t1, cw, score1);
  k_attn  <<<B_ * H_,  NT, 0, stream>>>(t, cw, cb, padd, score1, out);
}
